// gat_vae_do_49959059587678
// MI455X (gfx1250) — compile-verified, matrix operand order checked by reading, not run on hardware
//
#include <hip/hip_runtime.h>
#include <hip/hip_bf16.h>

typedef __attribute__((ext_vector_type(16))) _Float16 v16h;
typedef __attribute__((ext_vector_type(8)))  float    v8f;
typedef __attribute__((ext_vector_type(4)))  int      i32x4;

#define B_    4
#define N_    32
#define EMB_  512
#define FEAT_ 512
#define DO_   496

#define AS1 __attribute__((address_space(1)))
#define AS3 __attribute__((address_space(3)))

#if defined(__has_builtin)
#if __has_builtin(__builtin_amdgcn_global_load_async_to_lds_b128) && \
    __has_builtin(__builtin_amdgcn_s_wait_asynccnt)
#define USE_ASYNC_LDS 1
#endif
#endif
#ifndef USE_ASYNC_LDS
#define USE_ASYNC_LDS 0
#endif

// ---------------------------------------------------------------------------
// WMMA fragment helpers (layouts per CDNA5 ISA 7.12.2, wave32)
// A 16x32 f16:  lane l -> row m = (l&15); elems e<8 : K = 8*(l>>4)+e
//                                          e>=8: K = 16 + 8*(l>>4)+(e-8)
// C/D 16x16 f32: vgpr r, lane l -> m = r + 8*(l>>4), n = l&15
// ---------------------------------------------------------------------------

__device__ inline v16h load_a_frag_f32(const float* src, int lda, int row0, int k0) {
  int l  = threadIdx.x & 31;
  int m  = row0 + (l & 15);
  int kb = k0 + 8 * (l >> 4);
  const float* p = src + m * lda + kb;
  v16h a;
#pragma unroll
  for (int e = 0; e < 8; ++e) a[e] = (_Float16)p[e];
#pragma unroll
  for (int e = 0; e < 8; ++e) a[8 + e] = (_Float16)p[16 + e];
  return a;
}

__device__ inline v16h load_b_frag_f32(const float* src, int ldb, int k0, int col0) {
  int l  = threadIdx.x & 31;
  int n  = col0 + (l & 15);
  int kb = k0 + 8 * (l >> 4);
  v16h bf;
#pragma unroll
  for (int e = 0; e < 8; ++e) bf[e] = (_Float16)src[(kb + e) * ldb + n];
#pragma unroll
  for (int e = 0; e < 8; ++e) bf[8 + e] = (_Float16)src[(kb + 16 + e) * ldb + n];
  return bf;
}

__device__ inline void store_c_frag(float* dst, int ldc, int row0, int col0,
                                    const v8f& c, bool relu) {
  int l  = threadIdx.x & 31;
  int n  = col0 + (l & 15);
  int mb = row0 + 8 * (l >> 4);
#pragma unroll
  for (int r = 0; r < 8; ++r) {
    float v = c[r];
    if (relu) v = v > 0.f ? v : 0.f;
    dst[(mb + r) * ldc + n] = v;
  }
}

// Nontemporal + relu variant for the 130MB X_do stream
__device__ inline void store_c_frag_nt_relu(float* dst, int ldc, int row0, int col0,
                                            const v8f& c) {
  int l  = threadIdx.x & 31;
  int n  = col0 + (l & 15);
  int mb = row0 + 8 * (l >> 4);
#pragma unroll
  for (int r = 0; r < 8; ++r) {
    float v = c[r];
    v = v > 0.f ? v : 0.f;
    __builtin_nontemporal_store(v, &dst[(mb + r) * ldc + n]);
  }
}

// ---------------------------------------------------------------------------
// K1/K4: generic 128 x 512 x K GEMM, one 16x16 tile per wave, K stepped by 32
// ---------------------------------------------------------------------------
__global__ void gemm_wmma_kernel(const float* __restrict__ A,
                                 const float* __restrict__ Bm,
                                 float* __restrict__ C,
                                 int M, int Nn, int K, bool relu) {
  int wave   = (int)(blockIdx.x * blockDim.x + threadIdx.x) >> 5;
  int nwaves = (int)(gridDim.x * blockDim.x) >> 5;
  int mt_n = M / 16, nt_n = Nn / 16;
  for (int t = wave; t < mt_n * nt_n; t += nwaves) {
    int mt = t / nt_n, nt = t % nt_n;
    v8f c = {};
    for (int k0 = 0; k0 < K; k0 += 32) {
      v16h a = load_a_frag_f32(A, K, mt * 16, k0);
      v16h b = load_b_frag_f32(Bm, Nn, k0, nt * 16);
      c = __builtin_amdgcn_wmma_f32_16x16x32_f16(false, a, false, b,
                                                 (short)0, c, false, false);
    }
    store_c_frag(C, Nn, mt * 16, nt * 16, c, relu);
  }
}

// ---------------------------------------------------------------------------
// K2: GAT attention. Wave i handles row i (lane j = col). adj mask & strict
// lower; masked softmax; attn == causal_graph. posmask = bits of attn>0.
// ---------------------------------------------------------------------------
__global__ void attn_kernel(const float* __restrict__ feat,
                            const float* __restrict__ adj,
                            const float* __restrict__ asrc,
                            const float* __restrict__ adst,
                            float* __restrict__ attn_out,
                            float* __restrict__ causal_out,
                            unsigned* __restrict__ posmask) {
  int b    = blockIdx.x;
  int wave = threadIdx.x >> 5;   // row i
  int lane = threadIdx.x & 31;   // col j
  __shared__ float sdot[N_], ddot[N_];

  const float* frow = feat + (b * N_ + wave) * FEAT_;
  float ps = 0.f, pd = 0.f;
  for (int k = lane; k < FEAT_; k += 32) {
    float f = frow[k];
    ps += f * asrc[k];
    pd += f * adst[k];
  }
#pragma unroll
  for (int off = 16; off; off >>= 1) {
    ps += __shfl_xor(ps, off, 32);
    pd += __shfl_xor(pd, off, 32);
  }
  if (lane == 0) { sdot[wave] = ps; ddot[wave] = pd; }
  __syncthreads();

  int i = wave, j = lane;
  float sc = sdot[i] + ddot[j];
  sc = sc > 0.f ? sc : 0.2f * sc;                       // leaky_relu(0.2)
  bool m = (adj[(b * N_ + i) * N_ + j] > 0.f) && (j < i);
  float msc = m ? sc : -1e9f;
  float mx = msc;
#pragma unroll
  for (int off = 16; off; off >>= 1) {
    float o = __shfl_xor(mx, off, 32);
    mx = o > mx ? o : mx;
  }
  float p = __expf(msc - mx);
  float sum = p;
#pragma unroll
  for (int off = 16; off; off >>= 1) sum += __shfl_xor(sum, off, 32);
  int anym = m ? 1 : 0;
#pragma unroll
  for (int off = 16; off; off >>= 1) anym |= __shfl_xor(anym, off, 32);
  float av = (m && anym) ? (p / sum) : 0.f;

  attn_out[(b * N_ + i) * N_ + j]   = av;
  causal_out[(b * N_ + i) * N_ + j] = av;   // causal_graph == attn

  unsigned long long bal = __ballot(av > 0.f);
  if (lane == 0) posmask[b * N_ + i] = (unsigned)bal;
}

// ---------------------------------------------------------------------------
// K3: fU = relu(attn @ feat), per batch 32x32 @ 32x512, K=32 = single WMMA
// ---------------------------------------------------------------------------
__global__ void batched_gemm32_kernel(const float* __restrict__ A,
                                      const float* __restrict__ Bm,
                                      float* __restrict__ C, bool relu) {
  int wave = (int)(blockIdx.x * blockDim.x + threadIdx.x) >> 5; // 256 waves
  const int tiles_per_batch = 2 * (FEAT_ / 16);                 // 64
  int b = wave / tiles_per_batch;
  int t = wave % tiles_per_batch;
  if (b >= B_) return;                                          // never taken
  int mt = t & 1, nt = t >> 1;
  const float* Ab = A + b * N_ * N_;
  const float* Bb = Bm + b * N_ * FEAT_;
  v16h a  = load_a_frag_f32(Ab, N_, mt * 16, 0);
  v16h bf = load_b_frag_f32(Bb, FEAT_, 0, nt * 16);
  v8f c = {};
  c = __builtin_amdgcn_wmma_f32_16x16x32_f16(false, a, false, bf,
                                             (short)0, c, false, false);
  store_c_frag(C + b * N_ * FEAT_, FEAT_, mt * 16, nt * 16, c, relu);
}

// ---------------------------------------------------------------------------
// K4b: pack feat_dec into f16 B-fragment layout: bfrag[b][nt][lane][16]
// ---------------------------------------------------------------------------
__global__ void pack_bfrag_kernel(const float* __restrict__ featdec,
                                  _Float16* __restrict__ bfrag) {
  int idx = (int)(blockIdx.x * blockDim.x + threadIdx.x);  // B*32*32 = 4096
  if (idx >= B_ * 32 * 32) return;
  int lane = idx & 31;
  int nt   = (idx >> 5) & 31;
  int b    = idx >> 10;
  const float* src = featdec + b * N_ * FEAT_;
  int n  = nt * 16 + (lane & 15);
  int kb = 8 * (lane >> 4);
  _Float16* dst = bfrag + idx * 16;
#pragma unroll
  for (int e = 0; e < 8; ++e) dst[e]     = (_Float16)src[(kb + e) * FEAT_ + n];
#pragma unroll
  for (int e = 0; e < 8; ++e) dst[8 + e] = (_Float16)src[(kb + 16 + e) * FEAT_ + n];
}

// ---------------------------------------------------------------------------
// Map matrix element (m,k) -> A-fragment (tile, lane, elem)
// ---------------------------------------------------------------------------
__device__ inline void afrag_pos(int m, int k, int& mt, int& l, int& e) {
  mt = m >> 4;
  l  = (m & 15) | (((k >> 3) & 1) << 4);
  e  = ((k >> 4) << 3) | (k & 7);
}

// ---------------------------------------------------------------------------
// K5: main branch. inv(I - L) via forward substitution (unit lower
// triangular), e/s outputs, W_main -> f16 A-fragments.
// ---------------------------------------------------------------------------
__global__ void inv_main_kernel(const float* __restrict__ attn,
                                const float* __restrict__ eps_main,
                                const float* __restrict__ fce_w,
                                const float* __restrict__ fce_b,
                                const float* __restrict__ fcs_w,
                                const float* __restrict__ fcs_b,
                                float* __restrict__ e_out,
                                float* __restrict__ s_out,
                                _Float16* __restrict__ wmain_frag) {
  int b = blockIdx.x;
  int j = threadIdx.x;            // column owned by this lane
  __shared__ float L[N_][N_];
  for (int i = 0; i < N_; ++i) L[i][j] = attn[(b * N_ + i) * N_ + j];
  __syncthreads();

  float X[N_];
#pragma unroll
  for (int i = 0; i < N_; ++i) {
    float v = (i == j) ? 1.f : 0.f;
#pragma unroll
    for (int k = 0; k < i; ++k) v += L[i][k] * X[k];
    X[i] = v;
  }

  float few = fce_w[0], feb = fce_b[0], fsw = fcs_w[0], fsb = fcs_b[0];
#pragma unroll
  for (int i = 0; i < N_; ++i) {
    float inv = X[i];
    float e = few * inv + feb;
    float s = fsw * inv + fsb;
    e_out[(b * N_ + i) * N_ + j] = e;
    s_out[(b * N_ + i) * N_ + j] = s;
    float w = (inv == 0.f)
                  ? 0.f
                  : (e + __expf(0.5f * s) * eps_main[(b * N_ + i) * N_ + j]);
    int mt, l, ei;
    afrag_pos(i, j, mt, l, ei);
    wmain_frag[((b * 2 + mt) * 32 + l) * 16 + ei] = (_Float16)w;
  }
}

// ---------------------------------------------------------------------------
// K5b: X_raw = relu(W_main @ feat_dec) from prepacked fragments
// ---------------------------------------------------------------------------
__global__ void xraw_kernel(const _Float16* __restrict__ wmain_frag,
                            const _Float16* __restrict__ bfrag,
                            float* __restrict__ xraw) {
  int wave = (int)(blockIdx.x * blockDim.x + threadIdx.x) >> 5; // 256 waves
  int lane = threadIdx.x & 31;
  const int tiles_per_batch = 64;
  int b = wave / tiles_per_batch;
  int t = wave % tiles_per_batch;
  if (b >= B_) return;
  int mt = t & 1, nt = t >> 1;
  v16h a  = *(const v16h*)(wmain_frag + ((b * 2 + mt) * 32 + lane) * 16);
  v16h bf = *(const v16h*)(bfrag + ((b * 32 + nt) * 32 + lane) * 16);
  v8f c = {};
  c = __builtin_amdgcn_wmma_f32_16x16x32_f16(false, a, false, bf,
                                             (short)0, c, false, false);
  store_c_frag(xraw + b * N_ * FEAT_, FEAT_, mt * 16, nt * 16, c, true);
}

// ---------------------------------------------------------------------------
// K6: fused do-branch. One block per (b,d) of B*DO = 1984.
//  - async-stage this batch's 32KB B-fragment slice into LDS (ASYNCcnt path)
//  - label from bitmask logic (NT stores)
//  - wave0: forward substitution on pair-deleted graph -> W_do A-frags in LDS
//  - 8 waves x 8 WMMA tiles from LDS, NT-streamed X_do
// ---------------------------------------------------------------------------
__global__ void __launch_bounds__(256)
do_branch_kernel(const float* __restrict__ attn,
                 const unsigned* __restrict__ posmask,
                 const _Float16* __restrict__ bfrag,
                 const float* __restrict__ eps_do,
                 const float* __restrict__ fce_w,
                 const float* __restrict__ fce_b,
                 const float* __restrict__ fcs_w,
                 const float* __restrict__ fcs_b,
                 float* __restrict__ xdo,
                 float* __restrict__ label) {
  int bd = blockIdx.x;
  int b  = bd / DO_;
  int d  = bd % DO_;
  // triu_indices(N,1) pair (pa,pb)
  int rem = d, pa = 0;
  while (rem >= N_ - 1 - pa) { rem -= N_ - 1 - pa; ++pa; }
  int pb = pa + 1 + rem;

  __shared__ float L[N_][N_];
  __shared__ __align__(32) _Float16 afrag[2][32][16];
  __shared__ __align__(32) _Float16 btile[32][32][16];  // 32 KB batch B-frags

  int tid = threadIdx.x;
  const _Float16* bbase = bfrag + (long long)b * 32 * 32 * 16;

  // --- stage B-fragments global -> LDS (async on CDNA5, 16B per op) ---
#if USE_ASYNC_LDS
  {
    for (int o = tid; o < 2048; o += 256) {
      AS1 i32x4* gsrc = (AS1 i32x4*)(bbase + (size_t)o * 8);           // 16B
      AS3 i32x4* ldst = (AS3 i32x4*)(&btile[0][0][0] + (size_t)o * 8); // 16B
      __builtin_amdgcn_global_load_async_to_lds_b128(gsrc, ldst, 0, 0);
    }
  }
#else
  {
    const int4* gs = (const int4*)bbase;
    int4* ls = (int4*)&btile[0][0][0];
    for (int o = tid; o < 2048; o += 256) ls[o] = gs[o];
  }
#endif

  // L = attn with rows pa, pb zeroed (cgd = L + I, unit lower triangular)
  for (int t = tid; t < N_ * N_; t += blockDim.x) {
    int i = t >> 5, j = t & 31;
    float v = attn[(b * N_ + i) * N_ + j];
    if (i == pa || i == pb) v = 0.f;
    L[i][j] = v;
  }
  // label (pure bitmask logic, write-once stream -> NT stores)
  const unsigned* pm = posmask + b * N_;
  for (int t = tid; t < N_ * N_; t += blockDim.x) {
    int i = t >> 5, j = t & 31;
    bool lower = i > j;
    bool c23 = lower && (((j == pa) && (i != pb)) || ((j == pb) && (i != pa)));
    bool c4  = lower && (j != pa) && (j != pb) && (i != pa) && (i != pb);
    bool pos = (pm[i] >> j) & 1u;
    unsigned Pi = pm[i] & ((1u << i) - 1u);
    unsigned Pj = pm[j] & ((1u << j) - 1u);
    bool common = (Pi & Pj) != 0u;
    bool lab = (i == j) || (c23 && pos) || (c4 && common);
    __builtin_nontemporal_store(lab ? 1.f : 0.f,
                                &label[(long long)bd * (N_ * N_) + t]);
  }
  __syncthreads();

  // wave 0: triangular solve + reparameterization into A-fragments
  if (tid < 32) {
    int j = tid;
    float X[N_];
#pragma unroll
    for (int i = 0; i < N_; ++i) {
      float v = (i == j) ? 1.f : 0.f;
#pragma unroll
      for (int k = 0; k < i; ++k) v += L[i][k] * X[k];
      X[i] = v;
    }
    float few = fce_w[0], feb = fce_b[0], fsw = fcs_w[0], fsb = fcs_b[0];
#pragma unroll
    for (int i = 0; i < N_; ++i) {
      float inv = X[i];
      float w;
      if (inv == 0.f) {
        w = 0.f;
      } else {
        float e = few * inv + feb;
        float s = fsw * inv + fsb;
        float eps = __builtin_nontemporal_load(
            &eps_do[(long long)bd * (N_ * N_) + i * N_ + j]);
        w = e + __expf(0.5f * s) * eps;
      }
      int mt, l, ei;
      afrag_pos(i, j, mt, l, ei);
      afrag[mt][l][ei] = (_Float16)w;
    }
  }

  // all waves: drain own async copies, then block-wide barrier
#if USE_ASYNC_LDS
  __builtin_amdgcn_s_wait_asynccnt(0);
#endif
  __syncthreads();

  int wave = tid >> 5, lane = tid & 31;
  float* xout = xdo + (long long)bd * N_ * FEAT_;
#pragma unroll
  for (int ti = 0; ti < 8; ++ti) {
    int t  = wave * 8 + ti;       // 64 tiles over 8 waves
    int mt = t & 1, nt = t >> 1;
    v16h a  = *(const v16h*)(&afrag[mt][lane][0]);
    v16h bf = *(const v16h*)(&btile[nt][lane][0]);
    v8f c = {};
    c = __builtin_amdgcn_wmma_f32_16x16x32_f16(false, a, false, bf,
                                               (short)0, c, false, false);
    store_c_frag_nt_relu(xout, FEAT_, mt * 16, nt * 16, c);
  }
}

// ---------------------------------------------------------------------------
extern "C" void kernel_launch(void* const* d_in, const int* in_sizes, int n_in,
                              void* d_out, int out_size, void* d_ws, size_t ws_size,
                              hipStream_t stream) {
  const float* h        = (const float*)d_in[0];
  const float* adj      = (const float*)d_in[1];
  /* d_in[2] doc_len unused */
  const float* encWt    = (const float*)d_in[3];
  const float* asrc     = (const float*)d_in[4];
  const float* adst     = (const float*)d_in[5];
  const float* decWt    = (const float*)d_in[6];
  const float* fce_w    = (const float*)d_in[7];
  const float* fce_b    = (const float*)d_in[8];
  const float* fcs_w    = (const float*)d_in[9];
  const float* fcs_b    = (const float*)d_in[10];
  const float* eps_main = (const float*)d_in[11];
  const float* eps_do   = (const float*)d_in[12];

  char* ws = (char*)d_ws;
  float*    feat    = (float*)(ws + 0);        // 128x512 f32 = 256 KB
  float*    fU      = (float*)(ws + 262144);   // 256 KB
  float*    featdec = (float*)(ws + 524288);   // 256 KB
  float*    attn    = (float*)(ws + 786432);   // 16 KB
  unsigned* posmask = (unsigned*)(ws + 802816);// 512 B (padded)
  _Float16* bfrag   = (_Float16*)(ws + 803840);// 128 KB
  _Float16* wmfrag  = (_Float16*)(ws + 934912);// 8 KB

  float* out    = (float*)d_out;
  float* xraw   = out;                    // 4*32*512        = 65536
  float* xdo    = out + 65536;            // 1984*32*512     = 32505856
  float* label  = out + 32571392;         // 1984*32*32      = 2031616
  float* causal = out + 34603008;         // 4*1*32*32       = 4096
  float* e_out  = out + 34607104;         // 4096
  float* s_out  = out + 34611200;         // 4096

  // 1. feat = h @ enc_Wt
  gemm_wmma_kernel<<<32, 256, 0, stream>>>(h, encWt, feat, 128, FEAT_, EMB_, false);
  // 2. attention / softmax / causal_graph / posmask
  attn_kernel<<<B_, 1024, 0, stream>>>(feat, adj, asrc, adst, attn, causal, posmask);
  // 3. fU = relu(attn @ feat)
  batched_gemm32_kernel<<<32, 256, 0, stream>>>(attn, feat, fU, true);
  // 4. feat_dec = fU @ dec_Wt
  gemm_wmma_kernel<<<32, 256, 0, stream>>>(fU, decWt, featdec, 128, FEAT_, FEAT_, false);
  // 4b. pack feat_dec into WMMA B-fragments (f16)
  pack_bfrag_kernel<<<16, 256, 0, stream>>>(featdec, bfrag);
  // 5. main branch: inv / e / s / W_main fragments
  inv_main_kernel<<<B_, 32, 0, stream>>>(attn, eps_main, fce_w, fce_b, fcs_w, fcs_b,
                                         e_out, s_out, wmfrag);
  // 5b. X_raw = relu(W_main @ feat_dec)
  xraw_kernel<<<32, 256, 0, stream>>>(wmfrag, bfrag, xraw);
  // 6. fused do-branch: async LDS staging + label + triangular solve + X_do
  do_branch_kernel<<<B_ * DO_, 256, 0, stream>>>(attn, posmask, bfrag, eps_do,
                                                 fce_w, fce_b, fcs_w, fcs_b,
                                                 xdo, label);
}
